// ESTHead_5342939316490
// MI455X (gfx1250) — matrix operand order repaired and checked by reading, not compiled, not run on hardware
//
#include <hip/hip_runtime.h>
#include <hip/hip_bf16.h>
#include <math.h>

// ---------------------------------------------------------------------------
// Problem constants (from reference): B=4, S=2048, HID=4096, H=32, D=128
// ---------------------------------------------------------------------------
#define BSZ   4
#define SEQ   2048
#define HID   4096
#define NH    32
#define HD    128

// GEMM tiling
#define BM  256        // rows per block
#define BN  128        // == HD, so one N-tile == one head
#define BK  32         // == WMMA K for bf16
#define LDK 40         // padded LDS row stride (halves); 80B keeps 16B alignment
                       // (TDM: 16 dwords data + 4 dwords pad per row)

typedef __bf16 bf16;
typedef __attribute__((ext_vector_type(16))) __bf16 v16bf;
typedef __attribute__((ext_vector_type(8)))  __bf16 v8bf;
typedef __attribute__((ext_vector_type(8)))  float  v8f;
typedef __attribute__((ext_vector_type(4)))  unsigned int v4u;
typedef __attribute__((ext_vector_type(8)))  int v8i;
typedef __attribute__((ext_vector_type(4)))  int v4i;

#if __has_builtin(__builtin_amdgcn_tensor_load_to_lds)
#define USE_TDM 1
#else
#define USE_TDM 0
#endif

union Frag16 { v16bf v; v8bf h[2]; };

__device__ inline bf16 f2bf(float f) {
    union { float f; unsigned u; } x; x.f = f;
    unsigned r = (x.u + 0x7FFFu + ((x.u >> 16) & 1u)) >> 16;   // round-to-nearest-even
    unsigned short hs = (unsigned short)r;
    return __builtin_bit_cast(bf16, hs);
}

// inv_freq(j) = ROPE_BASE^(-2j/D) = exp(-j * 2*ln(10000)/128)
#define ROPE_C 0.14391156510f

// ---------------------------------------------------------------------------
// f32 -> bf16 conversion pass (run once per tensor).  n multiple of 8.
// ---------------------------------------------------------------------------
__global__ __launch_bounds__(256)
void cvt_bf16_kernel(const float* __restrict__ in, bf16* __restrict__ out, int n)
{
    int i = (blockIdx.x * 256 + threadIdx.x) * 8;
    if (i >= n) return;
    const float4* p = (const float4*)(in + i);
    float4 a = p[0], b = p[1];
    v8bf o;
    o[0] = f2bf(a.x); o[1] = f2bf(a.y); o[2] = f2bf(a.z); o[3] = f2bf(a.w);
    o[4] = f2bf(b.x); o[5] = f2bf(b.y); o[6] = f2bf(b.z); o[7] = f2bf(b.w);
    *(v8bf*)(out + i) = o;
}

// ---------------------------------------------------------------------------
// TDM descriptor group1 for a 2D bf16 tile (tileD0=BK elems wide, rows rows),
// row-major tensor with dim0 stride = HID elems, LDS padding 64B data + 16B.
// Field packing per CDNA5 ISA §8.4.
// ---------------------------------------------------------------------------
__device__ inline v8i tdm_group1(int rows, int tensorRows)
{
    v8i g;
    g[0] = (int)((1u << 16)            // data_size = 1 -> 2 bytes
               | (1u << 20)            // pad_enable
               | (3u << 22)            // pad_interval: 2^(3+1) = 16 dwords
               | (3u << 25));          // pad_amount: 3+1 = 4 dwords
    g[1] = (int)((unsigned)HID << 16); // tensor_dim0[15:0] (barrier addr = 0)
    g[2] = (int)((unsigned)tensorRows << 16);  // dim0 hi=0 | tensor_dim1[15:0]
    g[3] = (int)((unsigned)BK << 16);  // dim1 hi=0 | tile_dim0
    g[4] = rows;                       // tile_dim1 | tile_dim2=0
    g[5] = HID;                        // tensor_dim0_stride[31:0]
    g[6] = 0;
    g[7] = 0;
    return g;
}

// ---------------------------------------------------------------------------
// Fused projection GEMM:  C[m,n] = sum_k X[m,k] * W[n,k]   (bf16 WMMA, f32 acc)
// Inputs pre-converted bf16.  Double-buffered LDS tiles; staging done by the
// Tensor Data Mover (async DMA, TENSORcnt) when available, else VGPR copies.
//   WRITE_V=false: reduce q[h,d]*k -> scores  (K never touches HBM)
//   WRITE_V=true : write rope'd V (B,S,H,D) f32
// grid = (M/256, NH), block = 256 threads (8 waves: 4(m) x 2(n)),
// wave computes 64x64 = 4x4 WMMA tiles (128 accumulator VGPRs).
// ---------------------------------------------------------------------------
template<bool WRITE_V>
__global__ __launch_bounds__(256)
void gemm_kv_kernel(const bf16* __restrict__ Xb, const bf16* __restrict__ Wb,
                    const float* __restrict__ Q, float* __restrict__ scores,
                    float* __restrict__ vbuf)
{
    __shared__ bf16 As[2][BM * LDK];
    __shared__ bf16 Bs[2][BN * LDK];
    __shared__ float sS[BM];

    const int tid   = threadIdx.x;
    const int lane  = tid & 31;
    const int lo    = lane & 15;
    const int hi    = lane >> 4;
    const int wave  = tid >> 5;
    const int waveM = wave >> 1;     // 0..3 -> 64 rows each
    const int waveN = wave & 1;      // 0..1 -> 64 cols each

    const int head  = blockIdx.y;
    const int mBase = blockIdx.x * BM;
    const int nBase = head * HD;

    v8f acc[4][4];
#pragma unroll
    for (int i = 0; i < 4; ++i)
#pragma unroll
        for (int j = 0; j < 4; ++j)
            acc[i][j] = v8f{};

    const int ksteps = HID / BK;
    int buf = 0;

#if USE_TDM
    // ---- Tensor Data Mover staging: one 2D-tile DMA per matrix per k-step --
    const unsigned ldsA[2] = { (unsigned)(size_t)&As[0][0], (unsigned)(size_t)&As[1][0] };
    const unsigned ldsB[2] = { (unsigned)(size_t)&Bs[0][0], (unsigned)(size_t)&Bs[1][0] };
    const v8i g1x = tdm_group1(BM, BSZ * SEQ);
    const v8i g1w = tdm_group1(BN, NH * HD);
    const v4i z4 = {0, 0, 0, 0};
    const v8i z8 = {0, 0, 0, 0, 0, 0, 0, 0};

    auto issue_tdm = [&](int bsel, int kt) {
        const size_t gaX = (size_t)(Xb + (size_t)mBase * HID + (size_t)kt * BK);
        const size_t gaW = (size_t)(Wb + (size_t)nBase * HID + (size_t)kt * BK);
        v4u g0x = { 1u, ldsA[bsel], (unsigned)gaX,
                    ((unsigned)(gaX >> 32) & 0x01FFFFFFu) | 0x80000000u };  // type=2
        v4u g0w = { 1u, ldsB[bsel], (unsigned)gaW,
                    ((unsigned)(gaW >> 32) & 0x01FFFFFFu) | 0x80000000u };
        __builtin_amdgcn_tensor_load_to_lds(g0x, g1x, z4, z4, z8, 0);
        __builtin_amdgcn_tensor_load_to_lds(g0w, g1w, z4, z4, z8, 0);
    };

    if (wave == 0) issue_tdm(0, 0);
#else
    // ---- fallback: per-thread VGPR staging (pure bf16 copies) ----
    auto stage = [&](int bsel, int kt) {
        const int k0 = kt * BK;
        {
            const uint4* gx = (const uint4*)(Xb + (size_t)(mBase + tid) * HID + k0);
            uint4 t0 = gx[0], t1 = gx[1], t2 = gx[2], t3 = gx[3];
            uint4* la = (uint4*)&As[bsel][tid * LDK];
            la[0] = t0; la[1] = t1; la[2] = t2; la[3] = t3;
        }
        {
            const int r = tid >> 1, hf = tid & 1;
            const uint4* gw = (const uint4*)(Wb + (size_t)(nBase + r) * HID + k0 + hf * 16);
            uint4 t0 = gw[0], t1 = gw[1];
            uint4* lb = (uint4*)&Bs[bsel][r * LDK + hf * 16];
            lb[0] = t0; lb[1] = t1;
        }
    };
    stage(0, 0);
    __syncthreads();
#endif

    for (int kt = 0; kt < ksteps; ++kt) {
#if USE_TDM
        if (wave == 0) __builtin_amdgcn_s_wait_tensorcnt(0);
        __syncthreads();                       // tile `buf` ready for everyone
        if (wave == 0 && kt + 1 < ksteps) issue_tdm(buf ^ 1, kt + 1);
#else
        if (kt + 1 < ksteps) stage(buf ^ 1, kt + 1);
#endif

        // ---- WMMA fragments: contiguous k = 16*hi + s, same order in A and B ----
        Frag16 bfr[4];
#pragma unroll
        for (int ns = 0; ns < 4; ++ns) {
            const int r = waveN * 64 + ns * 16 + lo;
            bfr[ns].h[0] = *(const v8bf*)&Bs[buf][r * LDK + hi * 16];     // K[hi*16 .. +7]
            bfr[ns].h[1] = *(const v8bf*)&Bs[buf][r * LDK + hi * 16 + 8]; // K[hi*16+8 .. +15]
        }
#pragma unroll
        for (int ms = 0; ms < 4; ++ms) {
            Frag16 af;
            const int r = waveM * 64 + ms * 16 + lo;
            af.h[0] = *(const v8bf*)&As[buf][r * LDK + hi * 16];          // K[hi*16 .. +7]
            af.h[1] = *(const v8bf*)&As[buf][r * LDK + hi * 16 + 8];      // K[hi*16+8 .. +15]
#pragma unroll
            for (int ns = 0; ns < 4; ++ns)
                acc[ms][ns] = __builtin_amdgcn_wmma_f32_16x16x32_bf16(
                    false, af.v, false, bfr[ns].v,
                    (short)0, acc[ms][ns], false, false);
        }
        __syncthreads();                       // everyone done reading `buf`
        buf ^= 1;
    }

    // ------------------------- epilogue (RoPE) -----------------------------
    const int bIdx  = mBase >> 11;          // m / SEQ
    const int sBase = mBase & (SEQ - 1);

    if constexpr (WRITE_V) {
#pragma unroll
        for (int ms = 0; ms < 4; ++ms) {
#pragma unroll
            for (int ns = 0; ns < 4; ++ns) {
                const int d  = waveN * 64 + ns * 16 + lo;
                const float invf = __expf(-ROPE_C * (float)(d >> 1));
#pragma unroll
                for (int r = 0; r < 8; ++r) {
                    const int rowL = waveM * 64 + ms * 16 + hi * 8 + r;
                    const int m = mBase + rowL;
                    const int s = m & (SEQ - 1);
                    float sn, cs;
                    __sincosf((float)s * invf, &sn, &cs);
                    float x = acc[ms][ns][r];
                    float p = __shfl_xor(x, 1, 32);      // pair (d ^ 1)
                    float rp = (d & 1) ? (x * cs + p * sn) : (x * cs - p * sn);
                    vbuf[(size_t)m * HID + head * HD + d] = rp;
                }
            }
        }
    } else {
        sS[tid] = 0.0f;
        __syncthreads();
#pragma unroll
        for (int ms = 0; ms < 4; ++ms) {
            float rs[8];
#pragma unroll
            for (int r = 0; r < 8; ++r) rs[r] = 0.0f;
#pragma unroll
            for (int ns = 0; ns < 4; ++ns) {
                const int d  = waveN * 64 + ns * 16 + lo;
                const float invf = __expf(-ROPE_C * (float)(d >> 1));
                const float qv = Q[head * HD + d];
#pragma unroll
                for (int r = 0; r < 8; ++r) {
                    const int rowL = waveM * 64 + ms * 16 + hi * 8 + r;
                    const int s = sBase + rowL;
                    float sn, cs;
                    __sincosf((float)s * invf, &sn, &cs);
                    float x = acc[ms][ns][r];
                    float p = __shfl_xor(x, 1, 32);
                    float rp = (d & 1) ? (x * cs + p * sn) : (x * cs - p * sn);
                    rs[r] += qv * rp;
                }
            }
#pragma unroll
            for (int r = 0; r < 8; ++r) {
                float v = rs[r];
                v += __shfl_xor(v, 1, 32);
                v += __shfl_xor(v, 2, 32);
                v += __shfl_xor(v, 4, 32);
                v += __shfl_xor(v, 8, 32);
                if (lo == 0) {
                    const int rowL = waveM * 64 + ms * 16 + hi * 8 + r;
                    atomicAdd(&sS[rowL], v);
                }
            }
        }
        __syncthreads();
        scores[((size_t)(bIdx * NH + head)) * SEQ + sBase + tid] = sS[tid];
    }
}

// ---------------------------------------------------------------------------
// Softmax over S (in place).  grid = B*H rows, 256 threads.
// ---------------------------------------------------------------------------
__global__ __launch_bounds__(256)
void softmax_kernel(float* __restrict__ sc)
{
    __shared__ float red[256];
    const int tid = threadIdx.x;
    float* p = sc + (size_t)blockIdx.x * SEQ;

    float m = -3.0e38f;
    for (int i = tid; i < SEQ; i += 256) m = fmaxf(m, p[i]);
    red[tid] = m; __syncthreads();
    for (int s = 128; s > 0; s >>= 1) {
        if (tid < s) red[tid] = fmaxf(red[tid], red[tid + s]);
        __syncthreads();
    }
    m = red[0]; __syncthreads();

    float sum = 0.0f;
    for (int i = tid; i < SEQ; i += 256) {
        float e = __expf(p[i] - m);
        p[i] = e; sum += e;
    }
    red[tid] = sum; __syncthreads();
    for (int s = 128; s > 0; s >>= 1) {
        if (tid < s) red[tid] += red[tid + s];
        __syncthreads();
    }
    const float inv = 1.0f / red[0];
    for (int i = tid; i < SEQ; i += 256) p[i] *= inv;
}

// ---------------------------------------------------------------------------
// out[b, h*D+d] = sum_s attn[b,h,s] * v[b,s,h,d]   (bandwidth-bound)
// grid = B*H, 128 threads (one per d)
// ---------------------------------------------------------------------------
__global__ __launch_bounds__(128)
void attn_v_kernel(const float* __restrict__ attn, const float* __restrict__ v,
                   float* __restrict__ out)
{
    __shared__ float aw[128];
    const int bh = blockIdx.x;
    const int b  = bh >> 5;
    const int h  = bh & 31;
    const int d  = threadIdx.x;

    const float* ap = attn + (size_t)bh * SEQ;
    const float* vb = v + (size_t)b * SEQ * HID + h * HD;

    float acc = 0.0f;
    for (int s0 = 0; s0 < SEQ; s0 += 128) {
        aw[d] = ap[s0 + d];
        __syncthreads();
#pragma unroll 4
        for (int i = 0; i < 128; ++i)
            acc += aw[i] * vb[(size_t)(s0 + i) * HID + d];
        __syncthreads();
    }
    out[(size_t)b * HID + h * HD + d] = acc;
}

// ---------------------------------------------------------------------------
// Batch-4 GEMV: Y[b,n] = sum_k X[b,k]*W[n,k] (+bias).  One wave per row n.
// grid = N/8, 256 threads (8 waves).  Pure weight-bandwidth bound.
// ---------------------------------------------------------------------------
__global__ __launch_bounds__(256)
void gemv4_kernel(const float* __restrict__ X, const float* __restrict__ W,
                  const float* __restrict__ bias, float* __restrict__ Y,
                  int N, int K)
{
    const int wave = threadIdx.x >> 5;
    const int lane = threadIdx.x & 31;
    const int n = blockIdx.x * 8 + wave;
    if (n >= N) return;

    const float4* wr = (const float4*)(W + (size_t)n * K);
    const float4* x4 = (const float4*)X;
    const int K4 = K >> 2;

    float a0 = 0.f, a1 = 0.f, a2 = 0.f, a3 = 0.f;
    for (int i = lane; i < K4; i += 32) {
        float4 w  = wr[i];
        float4 xa = x4[i];
        float4 xb = x4[K4 + i];
        float4 xc = x4[2 * K4 + i];
        float4 xd = x4[3 * K4 + i];
        a0 += w.x * xa.x + w.y * xa.y + w.z * xa.z + w.w * xa.w;
        a1 += w.x * xb.x + w.y * xb.y + w.z * xb.z + w.w * xb.w;
        a2 += w.x * xc.x + w.y * xc.y + w.z * xc.z + w.w * xc.w;
        a3 += w.x * xd.x + w.y * xd.y + w.z * xd.z + w.w * xd.w;
    }
#pragma unroll
    for (int mask = 16; mask > 0; mask >>= 1) {
        a0 += __shfl_xor(a0, mask, 32);
        a1 += __shfl_xor(a1, mask, 32);
        a2 += __shfl_xor(a2, mask, 32);
        a3 += __shfl_xor(a3, mask, 32);
    }
    if (lane == 0) {
        const float bv = bias ? bias[n] : 0.0f;
        Y[(size_t)0 * N + n] = a0 + bv;
        Y[(size_t)1 * N + n] = a1 + bv;
        Y[(size_t)2 * N + n] = a2 + bv;
        Y[(size_t)3 * N + n] = a3 + bv;
    }
}

// ---------------------------------------------------------------------------
// LayerNorm over last dim (K=4096).  grid = 4 rows, 256 threads.
// ---------------------------------------------------------------------------
__global__ __launch_bounds__(256)
void layernorm_kernel(const float* __restrict__ X, const float* __restrict__ g,
                      const float* __restrict__ bb, float* __restrict__ Y, int K)
{
    __shared__ float s1[256], s2[256];
    const int tid = threadIdx.x;
    const float* x = X + (size_t)blockIdx.x * K;
    float* y = Y + (size_t)blockIdx.x * K;

    float sum = 0.f, sq = 0.f;
    for (int i = tid; i < K; i += 256) {
        float v = x[i]; sum += v; sq += v * v;
    }
    s1[tid] = sum; s2[tid] = sq; __syncthreads();
    for (int s = 128; s > 0; s >>= 1) {
        if (tid < s) { s1[tid] += s1[tid + s]; s2[tid] += s2[tid + s]; }
        __syncthreads();
    }
    const float mu  = s1[0] / (float)K;
    const float var = s2[0] / (float)K - mu * mu;
    const float inv = rsqrtf(var + 1e-5f);
    for (int i = tid; i < K; i += 256)
        y[i] = (x[i] - mu) * inv * g[i] + bb[i];
}

// ---------------------------------------------------------------------------
// a[i] = a[i] * silu(gate[i])
// ---------------------------------------------------------------------------
__global__ __launch_bounds__(256)
void silu_mul_kernel(float* __restrict__ a, const float* __restrict__ gate, int n)
{
    int i = blockIdx.x * 256 + threadIdx.x;
    if (i < n) {
        float gv = gate[i];
        a[i] = a[i] * (gv / (1.0f + __expf(-gv)));
    }
}

// ---------------------------------------------------------------------------
extern "C" void kernel_launch(void* const* d_in, const int* in_sizes, int n_in,
                              void* d_out, int out_size, void* d_ws, size_t ws_size,
                              hipStream_t stream)
{
    const float* X   = (const float*)d_in[0];   // (B,S,HID)
    const float* Q   = (const float*)d_in[1];   // (H,D)
    const float* Wk  = (const float*)d_in[2];   // (H*D, HID)
    const float* Wv  = (const float*)d_in[3];
    const float* Wo  = (const float*)d_in[4];   // (HID, H*D)
    const float* W0  = (const float*)d_in[5];   // (3*HID, HID)
    const float* W1  = (const float*)d_in[6];
    const float* W2  = (const float*)d_in[7];   // (HID, 3*HID)
    const float* Wsm = (const float*)d_in[8];   // (2048, HID)
    const float* bs  = (const float*)d_in[9];   // (2048,)
    const float* lng = (const float*)d_in[10];
    const float* lnb = (const float*)d_in[11];
    float* out = (float*)d_out;                 // (4, 2048)

    const int nX = BSZ * SEQ * HID;             // 33,554,432
    const int nW = NH * HD * HID;               // 16,777,216

    // workspace layout (byte offsets; bf16 scratch first, then f32 buffers)
    char* wsb = (char*)d_ws;
    bf16* Xb   = (bf16*)wsb;                                  // 64 MB
    bf16* Wb   = (bf16*)(wsb + (size_t)nX * sizeof(bf16));    // 32 MB (Wk then Wv)
    float* vbuf   = (float*)(wsb + (size_t)(nX + nW) * sizeof(bf16));  // 128 MB
    float* scores = vbuf   + (size_t)BSZ * SEQ * HID;         // B*H*S = 262,144
    float* aout   = scores + (size_t)BSZ * NH * SEQ;          // 16,384
    float* t0     = aout + BSZ * HID;
    float* t1     = t0   + BSZ * HID;
    float* ta     = t1   + BSZ * HID;            // a    (4 x 12288)
    float* tg     = ta   + BSZ * 3 * HID;        // gate (4 x 12288)
    float* t2     = tg   + BSZ * 3 * HID;
    float* t3     = t2   + BSZ * HID;

    const dim3 gemmGrid(BSZ * SEQ / BM, NH);     // (32, 32)

    // bf16 conversion passes (Wb is reused for Wk then Wv, stream-ordered)
    cvt_bf16_kernel<<<nX / (256 * 8), 256, 0, stream>>>(X,  Xb, nX);
    cvt_bf16_kernel<<<nW / (256 * 8), 256, 0, stream>>>(Wk, Wb, nW);
    // K projection + RoPE + q·k scores (K never written to memory)
    gemm_kv_kernel<false><<<gemmGrid, 256, 0, stream>>>(Xb, Wb, Q, scores, nullptr);
    // V projection + RoPE -> vbuf
    cvt_bf16_kernel<<<nW / (256 * 8), 256, 0, stream>>>(Wv, Wb, nW);
    gemm_kv_kernel<true ><<<gemmGrid, 256, 0, stream>>>(Xb, Wb, nullptr, nullptr, vbuf);
    // softmax over S, in place
    softmax_kernel<<<BSZ * NH, 256, 0, stream>>>(scores);
    // out[b, h*D+d] = sum_s attn * v
    attn_v_kernel<<<BSZ * NH, 128, 0, stream>>>(scores, vbuf, aout);
    // tail: batch-4 GEMVs + layernorms + SwiGLU
    gemv4_kernel<<<HID / 8, 256, 0, stream>>>(aout, Wo, nullptr, t0, HID, HID);
    layernorm_kernel<<<BSZ, 256, 0, stream>>>(t0, lng, lnb, t1, HID);
    gemv4_kernel<<<3 * HID / 8, 256, 0, stream>>>(t1, W0, nullptr, ta, 3 * HID, HID);
    gemv4_kernel<<<3 * HID / 8, 256, 0, stream>>>(t1, W1, nullptr, tg, 3 * HID, HID);
    silu_mul_kernel<<<(BSZ * 3 * HID + 255) / 256, 256, 0, stream>>>(ta, tg, BSZ * 3 * HID);
    gemv4_kernel<<<HID / 8, 256, 0, stream>>>(ta, W2, nullptr, t2, HID, 3 * HID);
    layernorm_kernel<<<BSZ, 256, 0, stream>>>(t2, lng, lnb, t3, HID);
    gemv4_kernel<<<2048 / 8, 256, 0, stream>>>(t3, Wsm, bs, out, 2048, HID);
}
